// MetaController_66082366816579
// MI455X (gfx1250) — compile-verified, matrix operand order checked by reading, not run on hardware
//
#include <hip/hip_runtime.h>
#include <hip/hip_bf16.h>
#include <math.h>

// ---------------------------------------------------------------------------
// MetaController fused pipeline for gfx1250 (MI455X).
//   - bf16 WMMA (v_wmma_f32_16x16x32_bf16) for all GEMMs
//   - decoder 2nd GEMM halved by pre-reducing w2 -> s2 weights
//   - persistent-workgroup GRU recurrence with h in LDS (zero-padded M=16)
//   - rolled WMMA K-loops to keep fragments in VGPRs (no scratch spills)
// ---------------------------------------------------------------------------

typedef __bf16 bf16_t;
typedef __attribute__((ext_vector_type(16))) __bf16 v16bf;
typedef __attribute__((ext_vector_type(8)))  float  v8f;

#define BB  4
#define NN  512
#define DD  512
#define HH  1024
#define RR  16
#define G3  (3*DD)     // 1536
#define TOK (BB*NN)    // 2048
#define DR  (DD*RR)    // 8192

__device__ __forceinline__ float sigm(float x) { return 1.0f / (1.0f + __expf(-x)); }

// Load one 16x32 bf16 WMMA fragment for this lane. `p` points at
// element [row][k0 + koff] of a row-major tile with row stride = K.
// Per ISA: elems 0..7 are K = k0+koff..+7, elems 8..15 are K = k0+16+koff..+7.
__device__ __forceinline__ v16bf load_frag(const bf16_t* p) {
    union { v16bf v; uint4 u[2]; } t;
    t.u[0] = *(const uint4*)(p);
    t.u[1] = *(const uint4*)(p + 16);
    return t.v;
}

// ---------------------------------------------------------------------------
// Generic WMMA GEMM:  C[M x N] = act( A[M x K] @ W[N x K]^T + bias[N] )
// A, W row-major bf16. Each wave computes a (16*MT) x (16*NT) tile.
// ACT: 0 = none, 1 = SiLU. Optionally mirrors C into bf16 (Cbf).
// ---------------------------------------------------------------------------
template<int MT, int NT, int ACT>
__global__ __launch_bounds__(256)
void wmma_gemm_kernel(const bf16_t* __restrict__ A,
                      const bf16_t* __restrict__ W,
                      const float*  __restrict__ bias,
                      float*        __restrict__ C,
                      bf16_t*       __restrict__ Cbf,
                      int M, int N, int K)
{
    const int lane = threadIdx.x & 31;
    const int wave = threadIdx.x >> 5;
    const int gw   = blockIdx.x * (blockDim.x >> 5) + wave;
    const int tilesN = N / (16 * NT);
    const int tilesM = M / (16 * MT);
    if (gw >= tilesM * tilesN) return;
    const int tn = (gw % tilesN) * (16 * NT);
    const int tm = (gw / tilesN) * (16 * MT);

    const int row  = lane & 15;
    const int koff = (lane >> 4) << 3;

    v8f acc[MT][NT] = {};

    #pragma unroll 1
    for (int k0 = 0; k0 < K; k0 += 32) {
        v16bf a[MT], b[NT];
        #pragma unroll
        for (int mi = 0; mi < MT; ++mi)
            a[mi] = load_frag(A + (size_t)(tm + mi*16 + row) * K + k0 + koff);
        #pragma unroll
        for (int ni = 0; ni < NT; ++ni)
            b[ni] = load_frag(W + (size_t)(tn + ni*16 + row) * K + k0 + koff);
        #pragma unroll
        for (int mi = 0; mi < MT; ++mi)
            #pragma unroll
            for (int ni = 0; ni < NT; ++ni)
                acc[mi][ni] = __builtin_amdgcn_wmma_f32_16x16x32_bf16(
                    false, a[mi], false, b[ni], (short)0, acc[mi][ni], false, false);
    }

    // D layout: lanes 0-15 -> M = vgpr, N = lane; lanes 16-31 -> M = vgpr+8.
    const int nlo = lane & 15;
    const int mhi = (lane >> 4) << 3;
    #pragma unroll
    for (int mi = 0; mi < MT; ++mi) {
        #pragma unroll
        for (int ni = 0; ni < NT; ++ni) {
            const int   ncol = tn + ni*16 + nlo;
            const float bv   = bias ? bias[ncol] : 0.0f;
            #pragma unroll
            for (int i = 0; i < 8; ++i) {
                float v = acc[mi][ni][i] + bv;
                if (ACT == 1) v = v * sigm(v);   // SiLU
                const size_t idx = (size_t)(tm + mi*16 + mhi + i) * N + ncol;
                C[idx] = v;
                if (Cbf) Cbf[idx] = (bf16_t)v;
            }
        }
    }
}

// ---------------------------------------------------------------------------
// f32 -> bf16 conversion
// ---------------------------------------------------------------------------
__global__ void f32_to_bf16_kernel(const float* __restrict__ src,
                                   bf16_t* __restrict__ dst, int n)
{
    int i = blockIdx.x * blockDim.x + threadIdx.x;
    if (i < n) dst[i] = (bf16_t)src[i];
}

// ---------------------------------------------------------------------------
// Pre-reduce second half of dec_w2 over d:  W2s[r][h] = sum_d w2[d*R+r][h]
// and b2s[r] = sum_d b2[DR + d*R + r].  (s2 = hid @ W2s^T + b2s)
// ---------------------------------------------------------------------------
__global__ void w2_reduce_kernel(const float* __restrict__ w2,
                                 const float* __restrict__ b2,
                                 bf16_t* __restrict__ w2s_bf,
                                 float*  __restrict__ b2s)
{
    const int t = blockIdx.x * blockDim.x + threadIdx.x;
    if (t < RR * HH) {
        const int r = t / HH, h = t % HH;
        float s = 0.0f;
        for (int d = 0; d < DD; ++d)
            s += w2[(size_t)(DR + d*RR + r) * HH + h];
        w2s_bf[t] = (bf16_t)s;
    }
    if (t < RR) {
        float s = 0.0f;
        for (int d = 0; d < DD; ++d)
            s += b2[DR + d*RR + t];
        b2s[t] = s;
    }
}

// ---------------------------------------------------------------------------
// Persistent GRU recurrence. grid = 2 blocks (ap / su), 1024 threads each.
// h kept in LDS as zero-padded 16 x 512 bf16 tile; per step:
//   hp = h @ w_hh^T : K-step outer loop, A-frag from LDS reused for 3 N-tiles
//   gates -> h_new, streamed to global (f32, optional bf16 mirror).
// ---------------------------------------------------------------------------
__global__ __launch_bounds__(1024)
void gru_recurrence_kernel(const float*  __restrict__ xpA, const bf16_t* __restrict__ whhA,
                           const float*  __restrict__ bhhA, float* __restrict__ hA,
                           bf16_t* __restrict__ hbfA,
                           const float*  __restrict__ xpB, const bf16_t* __restrict__ whhB,
                           const float*  __restrict__ bhhB, float* __restrict__ hB,
                           bf16_t* __restrict__ hbfB)
{
    const float* xp;  const bf16_t* whh; const float* bhh;
    float* hseq; bf16_t* hseqbf;
    if (blockIdx.x == 0) { xp = xpA; whh = whhA; bhh = bhhA; hseq = hA; hseqbf = hbfA; }
    else                 { xp = xpB; whh = whhB; bhh = bhhB; hseq = hB; hseqbf = hbfB; }

    __shared__ __align__(16) bf16_t h_bf[16 * DD];   // 16 KB (rows 4..15 stay zero)
    __shared__ float h_f[BB * DD];                   //  8 KB
    __shared__ float hp[BB * G3];                    // 24 KB

    const int tid = threadIdx.x;
    for (int i = tid; i < 16 * DD; i += 1024) h_bf[i] = (bf16_t)0.0f;
    for (int i = tid; i < BB * DD; i += 1024) h_f[i]  = 0.0f;
    __syncthreads();

    const int lane = tid & 31;
    const int wave = tid >> 5;
    const int row  = lane & 15;
    const int koff = (lane >> 4) << 3;
    const int nlo  = lane & 15;

    // this wave's 3 column tiles (of 96): rows of whh
    const bf16_t* wrow0 = whh + (size_t)((wave*3 + 0)*16 + row) * DD + koff;
    const bf16_t* wrow1 = whh + (size_t)((wave*3 + 1)*16 + row) * DD + koff;
    const bf16_t* wrow2 = whh + (size_t)((wave*3 + 2)*16 + row) * DD + koff;
    const bf16_t* arow  = &h_bf[row * DD + koff];

    for (int t = 0; t < NN; ++t) {
        // hp = h @ whh^T : A-frag loaded once per K-step, reused 3x
        v8f acc0 = {}, acc1 = {}, acc2 = {};
        #pragma unroll 1
        for (int k0 = 0; k0 < DD; k0 += 32) {
            v16bf a  = load_frag(arow + k0);
            v16bf b0 = load_frag(wrow0 + k0);
            v16bf b1 = load_frag(wrow1 + k0);
            v16bf b2 = load_frag(wrow2 + k0);
            acc0 = __builtin_amdgcn_wmma_f32_16x16x32_bf16(false, a, false, b0, (short)0, acc0, false, false);
            acc1 = __builtin_amdgcn_wmma_f32_16x16x32_bf16(false, a, false, b1, (short)0, acc1, false, false);
            acc2 = __builtin_amdgcn_wmma_f32_16x16x32_bf16(false, a, false, b2, (short)0, acc2, false, false);
        }
        if (lane < 16) {                       // rows 0..3 = valid batches
            const int col0 = (wave*3 + 0)*16 + nlo;
            const int col1 = (wave*3 + 1)*16 + nlo;
            const int col2 = (wave*3 + 2)*16 + nlo;
            #pragma unroll
            for (int i = 0; i < 4; ++i) {
                hp[i * G3 + col0] = acc0[i];
                hp[i * G3 + col1] = acc1[i];
                hp[i * G3 + col2] = acc2[i];
            }
        }
        __syncthreads();

        // gate update: 4 batches x 512 channels, 2 per thread
        for (int e = tid; e < BB * DD; e += 1024) {
            const int bb = e >> 9;
            const int d  = e & (DD - 1);
            const float* xr = xp + (size_t)(bb * NN + t) * G3;
            const float hr = hp[bb*G3 + d]        + bhh[d];
            const float hz = hp[bb*G3 + DD + d]   + bhh[DD + d];
            const float hn = hp[bb*G3 + 2*DD + d] + bhh[2*DD + d];
            const float r  = sigm(xr[d] + hr);
            const float z  = sigm(xr[DD + d] + hz);
            const float n  = tanhf(xr[2*DD + d] + r * hn);
            const float hprev = h_f[bb*DD + d];
            const float hnew  = (1.0f - z) * n + z * hprev;
            h_f[bb*DD + d]  = hnew;
            h_bf[bb*DD + d] = (bf16_t)hnew;
            const size_t gi = (size_t)(bb * NN + t) * DD + d;
            hseq[gi] = hnew;
            if (hseqbf) hseqbf[gi] = (bf16_t)hnew;
        }
        __syncthreads();
    }
}

// ---------------------------------------------------------------------------
// beta[b,n] = sigmoid( su_h[b,n,:] . beta_w )   (one wave per token)
// ---------------------------------------------------------------------------
__global__ void beta_kernel(const float* __restrict__ suh,
                            const float* __restrict__ bw,
                            float* __restrict__ beta)
{
    const int lane  = threadIdx.x & 31;
    const int wave  = threadIdx.x >> 5;
    const int token = blockIdx.x * (blockDim.x >> 5) + wave;
    if (token >= TOK) return;
    const float* hrow = suh + (size_t)token * DD;
    float acc = 0.0f;
    #pragma unroll
    for (int i = 0; i < 16; ++i)
        acc += hrow[lane + 32*i] * bw[lane + 32*i];
    #pragma unroll
    for (int off = 16; off > 0; off >>= 1)
        acc += __shfl_xor(acc, off, 32);
    if (lane == 0) beta[token] = sigm(acc);
}

// ---------------------------------------------------------------------------
// sampled = mean + noise*exp(0.5*logvar);  gated = scan(beta, (1-beta)*sampled)
// one thread per (b, d) channel, sequential over n.
// ---------------------------------------------------------------------------
__global__ void sample_scan_kernel(const float* __restrict__ proposed,
                                   const float* __restrict__ noise,
                                   const float* __restrict__ beta,
                                   float*  __restrict__ gated,
                                   bf16_t* __restrict__ gatedbf)
{
    const int ch = blockIdx.x * blockDim.x + threadIdx.x;
    if (ch >= BB * DD) return;
    const int bb = ch >> 9;
    const int d  = ch & (DD - 1);
    float h = 0.0f;
    #pragma unroll 1
    for (int t = 0; t < NN; ++t) {
        const int tok = bb * NN + t;
        const float be   = beta[tok];
        const float mean = proposed[(size_t)tok * (2*DD) + d];
        const float lv   = proposed[(size_t)tok * (2*DD) + DD + d];
        const float s    = mean + noise[(size_t)tok * DD + d] * __expf(0.5f * lv);
        h = be * h + (1.0f - be) * s;
        gated[(size_t)tok * DD + d]   = h;
        gatedbf[(size_t)tok * DD + d] = (bf16_t)h;
    }
}

// ---------------------------------------------------------------------------
// out = residual + gated * (w1 @ s2)   for one 256-token chunk.
// block = one token (512 threads), w1 row per thread (16 contiguous floats).
// ---------------------------------------------------------------------------
__global__ void combine_kernel(const float* __restrict__ w1c,
                               const float* __restrict__ s2,
                               const float* __restrict__ gated,
                               const float* __restrict__ resid,
                               float* __restrict__ out, int base)
{
    __shared__ float s2s[RR];
    const int token = base + blockIdx.x;
    const int d     = threadIdx.x;
    if (threadIdx.x < RR) s2s[threadIdx.x] = s2[(size_t)token * RR + threadIdx.x];
    __syncthreads();
    const float* wr = w1c + (size_t)blockIdx.x * DR + d * RR;
    float acc = 0.0f;
    #pragma unroll
    for (int r = 0; r < RR; ++r) acc += wr[r] * s2s[r];
    const size_t idx = (size_t)token * DD + d;
    out[idx] = resid[idx] + gated[idx] * acc;
}

// ---------------------------------------------------------------------------
extern "C" void kernel_launch(void* const* d_in, const int* in_sizes, int n_in,
                              void* d_out, int out_size, void* d_ws, size_t ws_size,
                              hipStream_t stream)
{
    (void)in_sizes; (void)n_in; (void)out_size; (void)ws_size;

    const float* resid    = (const float*)d_in[0];
    const float* noise    = (const float*)d_in[1];
    const float* ap_w_ih  = (const float*)d_in[2];
    const float* ap_w_hh  = (const float*)d_in[3];
    const float* ap_b_ih  = (const float*)d_in[4];
    const float* ap_b_hh  = (const float*)d_in[5];
    const float* ap_out_w = (const float*)d_in[6];
    const float* su_w_ih  = (const float*)d_in[7];
    const float* su_w_hh  = (const float*)d_in[8];
    const float* su_b_ih  = (const float*)d_in[9];
    const float* su_b_hh  = (const float*)d_in[10];
    const float* beta_w   = (const float*)d_in[11];
    const float* dec_w1   = (const float*)d_in[12];
    const float* dec_b1   = (const float*)d_in[13];
    const float* dec_w2   = (const float*)d_in[14];
    const float* dec_b2   = (const float*)d_in[15];
    float* out = (float*)d_out;

    char*  ws  = (char*)d_ws;
    size_t off = 0;
    auto alloc = [&](size_t bytes) -> void* {
        void* p = ws + off;
        off += (bytes + 255) & ~(size_t)255;
        return p;
    };

    bf16_t* x_bf      = (bf16_t*)alloc((size_t)TOK * DD * 2);
    bf16_t* apwih_bf  = (bf16_t*)alloc((size_t)G3 * DD * 2);
    bf16_t* apwhh_bf  = (bf16_t*)alloc((size_t)G3 * DD * 2);
    bf16_t* suwih_bf  = (bf16_t*)alloc((size_t)G3 * DD * 2);
    bf16_t* suwhh_bf  = (bf16_t*)alloc((size_t)G3 * DD * 2);
    bf16_t* apoutw_bf = (bf16_t*)alloc((size_t)2 * DD * DD * 2);
    bf16_t* decw1_bf  = (bf16_t*)alloc((size_t)HH * DD * 2);
    bf16_t* w1w_bf    = (bf16_t*)alloc((size_t)DR * HH * 2);   // first half of dec_w2
    bf16_t* w2s_bf    = (bf16_t*)alloc((size_t)RR * HH * 2);
    float*  b2s       = (float*) alloc((size_t)RR * 4);
    float*  ap_xp     = (float*) alloc((size_t)TOK * G3 * 4);
    float*  su_xp     = (float*) alloc((size_t)TOK * G3 * 4);
    float*  ap_h      = (float*) alloc((size_t)TOK * DD * 4);
    bf16_t* ap_h_bf   = (bf16_t*)alloc((size_t)TOK * DD * 2);
    float*  su_h      = (float*) alloc((size_t)TOK * DD * 4);
    float*  beta      = (float*) alloc((size_t)TOK * 4);
    float*  gated     = (float*) alloc((size_t)TOK * DD * 4);
    bf16_t* gated_bf  = (bf16_t*)alloc((size_t)TOK * DD * 2);
    bf16_t* hid_bf    = (bf16_t*)alloc((size_t)TOK * HH * 2);
    float*  s2        = (float*) alloc((size_t)TOK * RR * 4);
    float*  w1_chunk  = (float*) alloc((size_t)256 * DR * 4);
    float*  proposed  = ap_xp;   // alias: ap_xp dead after recurrence
    float*  hid       = su_xp;   // alias: su_xp dead after recurrence

    // 1. precision conversions + w2 pre-reduction
    auto conv = [&](const float* s, bf16_t* d, int n) {
        f32_to_bf16_kernel<<<(n + 255) / 256, 256, 0, stream>>>(s, d, n);
    };
    conv(resid,    x_bf,      TOK * DD);
    conv(ap_w_ih,  apwih_bf,  G3 * DD);
    conv(ap_w_hh,  apwhh_bf,  G3 * DD);
    conv(su_w_ih,  suwih_bf,  G3 * DD);
    conv(su_w_hh,  suwhh_bf,  G3 * DD);
    conv(ap_out_w, apoutw_bf, 2 * DD * DD);
    conv(dec_w1,   decw1_bf,  HH * DD);
    conv(dec_w2,   w1w_bf,    DR * HH);            // first 8192 rows == w1 block
    w2_reduce_kernel<<<(RR * HH + 255) / 256, 256, 0, stream>>>(dec_w2, dec_b2, w2s_bf, b2s);

    // 2. GRU input projections: xp = x @ w_ih^T + b_ih  (M=2048, N=1536, K=512)
    wmma_gemm_kernel<2,2,0><<<(64*48 + 7)/8, 256, 0, stream>>>(
        x_bf, apwih_bf, ap_b_ih, ap_xp, nullptr, TOK, G3, DD);
    wmma_gemm_kernel<2,2,0><<<(64*48 + 7)/8, 256, 0, stream>>>(
        x_bf, suwih_bf, su_b_ih, su_xp, nullptr, TOK, G3, DD);

    // 3. both GRU recurrences concurrently (2 persistent workgroups)
    gru_recurrence_kernel<<<2, 1024, 0, stream>>>(
        ap_xp, apwhh_bf, ap_b_hh, ap_h, ap_h_bf,
        su_xp, suwhh_bf, su_b_hh, su_h, nullptr);

    // 4. proposed = ap_h @ ap_out_w^T   (M=2048, N=1024, K=512)
    wmma_gemm_kernel<2,2,0><<<(64*32 + 7)/8, 256, 0, stream>>>(
        ap_h_bf, apoutw_bf, nullptr, proposed, nullptr, TOK, 2*DD, DD);

    // 5. beta = sigmoid(su_h . beta_w)
    beta_kernel<<<TOK / 4, 128, 0, stream>>>(su_h, beta_w, beta);

    // 6. sample + associative gating scan -> gated
    sample_scan_kernel<<<(BB*DD) / 256, 256, 0, stream>>>(
        proposed, noise, beta, gated, gated_bf);

    // 7. hid = silu(gated @ dec_w1^T + b1)   (M=2048, N=1024, K=512)
    wmma_gemm_kernel<2,2,1><<<(64*32 + 7)/8, 256, 0, stream>>>(
        gated_bf, decw1_bf, dec_b1, hid, hid_bf, TOK, HH, DD);

    // 8. s2 = hid @ W2s^T + b2s   (M=2048, N=16, K=1024)
    wmma_gemm_kernel<2,1,0><<<(64*1 + 7)/8, 256, 0, stream>>>(
        hid_bf, w2s_bf, b2s, s2, nullptr, TOK, RR, HH);

    // 9. chunked w1 GEMM + fused control/output (8 x 256 tokens)
    for (int c = 0; c < 8; ++c) {
        const int base = c * 256;
        wmma_gemm_kernel<2,2,0><<<(8*256 + 7)/8, 256, 0, stream>>>(
            hid_bf + (size_t)base * HH, w1w_bf, dec_b2,
            w1_chunk, nullptr, 256, DR, HH);
        combine_kernel<<<256, 512, 0, stream>>>(w1_chunk, s2, gated, resid, out, base);
    }

    // 10. vae_kl_loss scalar = 0
    hipMemsetAsync(out + (size_t)TOK * DD, 0, sizeof(float), stream);
}
